// deformable_LKAConv_87076166959782
// MI455X (gfx1250) — compile-verified, matrix operand order checked by reading, not run on hardware
//
#include <hip/hip_runtime.h>

// ---------------------------------------------------------------------------
// Deformable LKA on MI455X (gfx1250).
//  - offset convs + pointwise as bf16 WMMA GEMMs (v_wmma_f32_16x16x32_bf16)
//  - fragment-ready B layout -> 2x global_load_b128 per chunk (clause'd)
//  - zero-padded (halo=9) NHWC bf16 activations -> branch-free A loads with
//    wave-uniform chunk offset (SALU only)
//  - 64-pixel M-tile: each B fragment feeds 4 WMMAs; 1 barrier per K-chunk
//  - ping-pong LDS A staging; global prefetch of next B fragment
// ---------------------------------------------------------------------------

typedef __attribute__((ext_vector_type(8)))  float  v8f;
typedef __attribute__((ext_vector_type(16))) __bf16 v16bf;

#define C_DIM 64
#define H_DIM 56
#define W_DIM 56
#define HW_DIM (H_DIM * W_DIM)
#define B_DIM 4
#define M_DIM (B_DIM * HW_DIM)   // 12544, divisible by 64

#define PW_HALO 9
#define WP_DIM (W_DIM + 2 * PW_HALO)   // 74
#define HP_DIM (H_DIM + 2 * PW_HALO)   // 74
#define PAD_ELEMS ((size_t)B_DIM * HP_DIM * WP_DIM * C_DIM)   // 1,401,856

__device__ __forceinline__ unsigned short f2bf(float f) {
    unsigned int u = __builtin_bit_cast(unsigned int, f);
    unsigned int r = u + 0x7FFFu + ((u >> 16) & 1u);   // round-to-nearest-even
    return (unsigned short)(r >> 16);
}

// ---------------------------------------------------------------------------
// NCHW f32 -> NHWC f32 (for bilinear gathers)
// ---------------------------------------------------------------------------
__global__ void k_nchw_to_nhwc(const float* __restrict__ X,
                               float* __restrict__ Xh, int total) {
    int i = blockIdx.x * blockDim.x + threadIdx.x;
    if (i >= total) return;
    int b = i / (C_DIM * HW_DIM);
    int r = i % (C_DIM * HW_DIM);
    int c = r / HW_DIM;
    int s = r % HW_DIM;
    Xh[(size_t)(b * HW_DIM + s) * C_DIM + c] = X[i];
}

// NCHW f32 -> padded NHWC bf16 (zero halo), one thread per padded element
__global__ void k_fill_xpad(const float* __restrict__ X,
                            unsigned short* __restrict__ Xp) {
    size_t i = (size_t)blockIdx.x * blockDim.x + threadIdx.x;
    if (i >= PAD_ELEMS) return;
    int c  = (int)(i & 63);
    int t  = (int)(i >> 6);
    int xp = t % WP_DIM;  t /= WP_DIM;
    int yp = t % HP_DIM;
    int b  = t / HP_DIM;
    int y = yp - PW_HALO, x = xp - PW_HALO;
    unsigned short v = 0;
    if (y >= 0 && y < H_DIM && x >= 0 && x < W_DIM)
        v = f2bf(X[((size_t)(b * C_DIM + c) * H_DIM + y) * W_DIM + x]);
    Xp[i] = v;
}

// zero a padded bf16 buffer (as u32), run before samplers fill the interior
__global__ void k_zero_u32(unsigned int* __restrict__ P, int n32) {
    int i = blockIdx.x * blockDim.x + threadIdx.x;
    if (i < n32) P[i] = 0u;
}

// ---------------------------------------------------------------------------
// Weights (COUT,64,K,K) f32 -> fragment-ready bf16 B:
//   Bf[((nt*NCHUNK + kc)*32 + lane)*8 + v] = pack(B[kg][n], B[kg+1][n])
// n = nt*16 + (lane&15), kb = (v>>2)*16 + (lane>>4)*8 + (v&3)*2,
// kg = kc*32 + kb, k-index = tap*64 + cin. Cols >= COUT zeroed.
// ---------------------------------------------------------------------------
template <int K, int NCOLS, int COUT>
__global__ void k_prep_bfrag(const float* __restrict__ Wsrc,
                             unsigned int* __restrict__ Bf) {
    constexpr int KDIM   = K * K * C_DIM;
    constexpr int NCHUNK = KDIM / 32;
    constexpr int NTILES = NCOLS / 16;
    int i = blockIdx.x * blockDim.x + threadIdx.x;
    if (i >= NTILES * NCHUNK * 32 * 8) return;
    const int v    = i & 7;
    const int lane = (i >> 3) & 31;
    const int kc   = (i >> 8) % NCHUNK;
    const int nt   = (i >> 8) / NCHUNK;
    const int n    = nt * 16 + (lane & 15);
    const int kb   = ((v >> 2) * 16) + ((lane >> 4) * 8) + ((v & 3) * 2);
    const int kg   = kc * 32 + kb;
    const int tap  = kg >> 6, ci = kg & 63;
    unsigned int lo = 0, hi = 0;
    if (n < COUT) {
        lo = f2bf(Wsrc[((size_t)(n * C_DIM + ci)) * (K * K) + tap]);
        hi = f2bf(Wsrc[((size_t)(n * C_DIM + ci + 1)) * (K * K) + tap]);
    }
    Bf[i] = lo | (hi << 16);
}

// depthwise weights (C,1,K,K) -> [tap][c] so sampler reads are coalesced
template <int K>
__global__ void k_prep_dwT(const float* __restrict__ Dw, float* __restrict__ DwT) {
    int i = blockIdx.x * blockDim.x + threadIdx.x;
    if (i >= K * K * C_DIM) return;
    int tap = i >> 6, c = i & 63;
    DwT[i] = Dw[(size_t)c * (K * K) + tap];
}

// ---------------------------------------------------------------------------
// Fused im2col + WMMA GEMM convolution over padded NHWC bf16 activations.
// Block = 128 threads (4 waves). Block owns a 64-pixel M-tile (4 subtiles of
// 16 rows sharing each B fragment -> 4 WMMAs per chunk per wave); wave w owns
// N-tile blockIdx.y*4+w. K-chunk = 32 (half the channels of one tap); the
// chunk address delta is wave-uniform, A loads are branch-free.
// ---------------------------------------------------------------------------
template <int K, int PAD, int DIL, int NCOLS, int COUT>
__launch_bounds__(128)
__global__ void k_wmma_conv(const unsigned short* __restrict__ Apad, // padded NHWC bf16
                            const uint4* __restrict__ Bfrag,         // fragment-ready
                            const float* __restrict__ bias,
                            float* __restrict__ Out) {               // [M][NCOLS]
    constexpr int MSUB   = 4;
    constexpr int MT     = 16 * MSUB;            // 64 pixels per block
    constexpr int KDIM   = K * K * C_DIM;
    constexpr int NCHUNK = KDIM / 32;
    constexpr int NTILES = NCOLS / 16;

    __shared__ unsigned short As[2][MT][36];     // ping-pong, rows padded to 72B

    const int lane  = threadIdx.x & 31;
    const int wave  = threadIdx.x >> 5;
    const int mtile = blockIdx.x;
    const int ntile = blockIdx.y * 4 + wave;
    const int ntl   = (ntile < NTILES) ? ntile : 0;   // clamp idle wave

    // A-staging: thread loads pixels (pa + 16h), channels [ja*4 .. ja*4+3]
    const int pa = threadIdx.x >> 3;
    const int ja = threadIdx.x & 7;
    int pixbase[MSUB];
#pragma unroll
    for (int h = 0; h < MSUB; ++h) {
        const int ma = mtile * MT + pa + 16 * h;
        const int b  = ma / HW_DIM;
        const int sa = ma % HW_DIM;
        const int y  = sa / W_DIM, x = sa % W_DIM;
        pixbase[h] = ((b * HP_DIM + y + PW_HALO) * WP_DIM + (x + PW_HALO)) * C_DIM + ja * 4;
    }

    // fragment coords
    const int fn       = lane & 15;
    const int fg       = lane >> 4;
    const int n_global = ntile * 16 + fn;

    const uint4* bptr = Bfrag + (size_t)ntl * NCHUNK * 64 + lane * 2;

    auto gload = [&](int kc, uint2 (&regs)[MSUB]) {
        const int tap  = kc >> 1;
        const int c0   = (kc & 1) * 32;
        const int ky   = tap / K, kx = tap % K;
        const int doff = ((ky * DIL - PAD) * WP_DIM + (kx * DIL - PAD)) * C_DIM + c0; // uniform
#pragma unroll
        for (int h = 0; h < MSUB; ++h)
            regs[h] = *(const uint2*)(Apad + (pixbase[h] + doff));
    };

    v8f acc[MSUB];
#pragma unroll
    for (int h = 0; h < MSUB; ++h) acc[h] = (v8f){};

    uint2 regs[MSUB];
    gload(0, regs);

    for (int kc = 0; kc < NCHUNK; ++kc) {
        const int buf = kc & 1;
        // ping-pong: readers of this buffer were fenced by barrier in kc-1
#pragma unroll
        for (int h = 0; h < MSUB; ++h) {
            *(unsigned int*)&As[buf][pa + 16 * h][ja * 4]     = regs[h].x;
            *(unsigned int*)&As[buf][pa + 16 * h][ja * 4 + 2] = regs[h].y;
        }
        if (kc + 1 < NCHUNK) {
            gload(kc + 1, regs);                               // overlap w/ barrier
            __builtin_prefetch(bptr + (size_t)(kc + 1) * 64, 0, 3);
        }
        __syncthreads();

        union FragB { uint4 q[2]; v16bf v; } bfr;
        bfr.q[0] = bptr[(size_t)kc * 64];
        bfr.q[1] = bptr[(size_t)kc * 64 + 1];

#pragma unroll
        for (int h = 0; h < MSUB; ++h) {
            union FragA { unsigned int u[8]; v16bf v; } af;
#pragma unroll
            for (int v = 0; v < 8; ++v) {
                const int kb = ((v >> 2) * 16) + (fg * 8) + ((v & 3) * 2);
                af.u[v] = *(const unsigned int*)&As[buf][fn + 16 * h][kb];
            }
            acc[h] = __builtin_amdgcn_wmma_f32_16x16x32_bf16(
                false, af.v, false, bfr.v, (short)0, acc[h], false, false);
        }
    }

    if (ntile < NTILES && n_global < COUT) {
        const float bv = bias[n_global];
#pragma unroll
        for (int h = 0; h < MSUB; ++h) {
#pragma unroll
            for (int r = 0; r < 8; ++r) {
                const int mrow = 16 * h + fg * 8 + r;   // C/D layout: lanes 16-31 -> M+8
                Out[(size_t)(mtile * MT + mrow) * NCOLS + n_global] = acc[h][r] + bv;
            }
        }
    }
}

// ---------------------------------------------------------------------------
// Deformable depthwise sampler. Block = 64 threads = one output pixel,
// thread = channel. Offsets staged once in LDS; NHWC corner gathers are
// 256B-contiguous per corner. Writes f32 (unpadded NHWC) + bf16 (padded NHWC
// interior; halo pre-zeroed by k_zero_u32).
// ---------------------------------------------------------------------------
template <int K, int PAD, int DIL, int OFFSTRIDE>
__launch_bounds__(64)
__global__ void k_deform_dw(const float* __restrict__ Xh,     // NHWC f32
                            const float* __restrict__ Off,    // [M][OFFSTRIDE]
                            const float* __restrict__ DwT,    // [K*K][64]
                            float* __restrict__ OutF,         // NHWC f32
                            unsigned short* __restrict__ OutBp) { // padded NHWC bf16
    __shared__ float offs[2 * K * K];
    const int m = blockIdx.x;
    const int c = threadIdx.x;
    const int b = m / HW_DIM, s = m % HW_DIM;
    const int y = s / W_DIM, x = s % W_DIM;

    for (int j = threadIdx.x; j < 2 * K * K; j += 64)
        offs[j] = Off[(size_t)m * OFFSTRIDE + j];
    __syncthreads();

    float acc = 0.f;
#pragma unroll 1
    for (int tap = 0; tap < K * K; ++tap) {
        const int ky = tap / K, kx = tap % K;
        const float py = (float)(y + ky * DIL - PAD) + offs[2 * tap];
        const float px = (float)(x + kx * DIL - PAD) + offs[2 * tap + 1];
        const float y0 = floorf(py), x0 = floorf(px);
        const float wy1 = py - y0, wx1 = px - x0;
        const int iy0 = (int)y0, ix0 = (int)x0;

        float sample = 0.f;
#pragma unroll
        for (int dy = 0; dy < 2; ++dy) {
#pragma unroll
            for (int dx = 0; dx < 2; ++dx) {
                const int yy = iy0 + dy, xx = ix0 + dx;
                const float wgt = (dy ? wy1 : 1.f - wy1) * (dx ? wx1 : 1.f - wx1);
                if (yy >= 0 && yy < H_DIM && xx >= 0 && xx < W_DIM)
                    sample += wgt * Xh[((size_t)(b * HW_DIM + yy * W_DIM + xx)) * C_DIM + c];
            }
        }
        acc += DwT[tap * C_DIM + c] * sample;
    }
    OutF[(size_t)m * C_DIM + c] = acc;
    OutBp[((size_t)(b * HP_DIM + y + PW_HALO) * WP_DIM + (x + PW_HALO)) * C_DIM + c] = f2bf(acc);
}

// ---------------------------------------------------------------------------
// out(NCHW) = x(NCHW) * pw(NHWC)
// ---------------------------------------------------------------------------
__global__ void k_final_mul(const float* __restrict__ X,
                            const float* __restrict__ Pw,
                            float* __restrict__ Out, int total) {
    int i = blockIdx.x * blockDim.x + threadIdx.x;
    if (i >= total) return;
    int b = i / (C_DIM * HW_DIM);
    int r = i % (C_DIM * HW_DIM);
    int c = r / HW_DIM;
    int s = r % HW_DIM;
    Out[i] = X[i] * Pw[(size_t)(b * HW_DIM + s) * C_DIM + c];
}

// ---------------------------------------------------------------------------
static inline size_t align256(size_t x) { return (x + 255) & ~(size_t)255; }

extern "C" void kernel_launch(void* const* d_in, const int* in_sizes, int n_in,
                              void* d_out, int out_size, void* d_ws, size_t ws_size,
                              hipStream_t stream) {
    const float* x      = (const float*)d_in[0];
    const float* off_w1 = (const float*)d_in[1];
    const float* off_b1 = (const float*)d_in[2];
    const float* dw_w1  = (const float*)d_in[3];
    const float* off_w2 = (const float*)d_in[4];
    const float* off_b2 = (const float*)d_in[5];
    const float* dw_w2  = (const float*)d_in[6];
    const float* pw_w   = (const float*)d_in[7];
    const float* pw_b   = (const float*)d_in[8];
    float* out = (float*)d_out;

    // ---- workspace carve-out (~31 MB) ----
    char* p = (char*)d_ws;
    auto carve = [&](size_t bytes) { char* q = p; p += align256(bytes); return q; };
    float*          Xh   = (float*)         carve((size_t)M_DIM * 64 * 4);
    unsigned short* Xbp  = (unsigned short*)carve(PAD_ELEMS * 2);
    unsigned int*   Bf1  = (unsigned int*)  carve((size_t)4 * 50 * 256 * 4);   // 200 KB
    float*          DwT1 = (float*)         carve((size_t)25 * 64 * 4);
    float*          Off1 = (float*)         carve((size_t)M_DIM * 64 * 4);
    float*          A1f  = (float*)         carve((size_t)M_DIM * 64 * 4);
    unsigned short* A1bp = (unsigned short*)carve(PAD_ELEMS * 2);
    unsigned int*   Bf2  = (unsigned int*)  carve((size_t)7 * 98 * 256 * 4);   // 686 KB
    float*          DwT2 = (float*)         carve((size_t)49 * 64 * 4);
    float*          Off2 = (float*)         carve((size_t)M_DIM * 112 * 4);
    float*          A2f  = (float*)         carve((size_t)M_DIM * 64 * 4);     // dummy sink
    unsigned short* A2bp = (unsigned short*)carve(PAD_ELEMS * 2);
    unsigned int*   Bfp  = (unsigned int*)  carve((size_t)4 * 2 * 256 * 4);    // 8 KB
    float*          Pwo  = (float*)         carve((size_t)M_DIM * 64 * 4);

    const int total   = B_DIM * C_DIM * HW_DIM;     // 802816
    const int pad_u32 = (int)(PAD_ELEMS / 2);       // 700928

    // 0) layouts: NHWC f32 view + padded bf16 view of x; zero padded attn bufs
    k_nchw_to_nhwc<<<(total + 255) / 256, 256, 0, stream>>>(x, Xh, total);
    k_fill_xpad<<<(int)((PAD_ELEMS + 255) / 256), 256, 0, stream>>>(x, Xbp);
    k_zero_u32<<<(pad_u32 + 255) / 256, 256, 0, stream>>>((unsigned int*)A1bp, pad_u32);
    k_zero_u32<<<(pad_u32 + 255) / 256, 256, 0, stream>>>((unsigned int*)A2bp, pad_u32);

    // weight repacks
    k_prep_bfrag<5, 64, 50><<<(4 * 50 * 256 + 255) / 256, 256, 0, stream>>>(off_w1, Bf1);
    k_prep_dwT<5><<<(25 * 64 + 255) / 256, 256, 0, stream>>>(dw_w1, DwT1);
    k_prep_bfrag<7, 112, 98><<<(7 * 98 * 256 + 255) / 256, 256, 0, stream>>>(off_w2, Bf2);
    k_prep_dwT<7><<<(49 * 64 + 255) / 256, 256, 0, stream>>>(dw_w2, DwT2);
    k_prep_bfrag<1, 64, 64><<<(4 * 2 * 256 + 255) / 256, 256, 0, stream>>>(pw_w, Bfp);

    // ---- stage 1: K=5, pad=2, dil=1, Cout=50 (pad to 64) ----
    k_wmma_conv<5, 2, 1, 64, 50><<<dim3(M_DIM / 64, 1), 128, 0, stream>>>(
        Xbp, (const uint4*)Bf1, off_b1, Off1);
    k_deform_dw<5, 2, 1, 64><<<M_DIM, 64, 0, stream>>>(Xh, Off1, DwT1, A1f, A1bp);

    // ---- stage 2: K=7, pad=9, dil=3, Cout=98 (pad to 112) ----
    k_wmma_conv<7, 9, 3, 112, 98><<<dim3(M_DIM / 64, 2), 128, 0, stream>>>(
        A1bp, (const uint4*)Bf2, off_b2, Off2);
    k_deform_dw<7, 9, 3, 112><<<M_DIM, 64, 0, stream>>>(A1f, Off2, DwT2, A2f, A2bp);

    // ---- stage 3: pointwise (K=1) + gating ----
    k_wmma_conv<1, 0, 1, 64, 64><<<dim3(M_DIM / 64, 1), 128, 0, stream>>>(
        A2bp, (const uint4*)Bfp, pw_b, Pwo);
    k_final_mul<<<(total + 255) / 256, 256, 0, stream>>>(x, Pwo, out, total);
}
